// MultiMamba_74466142978813
// MI455X (gfx1250) — matrix operand order repaired and checked by reading, not compiled, not run on hardware
//
#include <hip/hip_runtime.h>
#include <hip/hip_bf16.h>

#define B_  32
#define L_  256
#define DM  256
#define DI  512
#define NS  16   // state dim N
#define RR  16   // dt rank R
#define KC  4    // conv kernel
#define ND  4    // directions
#define AR  64   // attn rank

typedef __attribute__((ext_vector_type(16))) __bf16 v16bf;
typedef __attribute__((ext_vector_type(8)))  float  v8f;
typedef __attribute__((ext_vector_type(4)))  unsigned uint32x4;
typedef __attribute__((ext_vector_type(8)))  int      int32x8;
typedef __attribute__((ext_vector_type(4)))  int      int32x4;

// native f32 -> bf16 conversion (lowers to v_cvt_pk_bf16_f32)
static __device__ __forceinline__ __bf16 f2bf(float f) { return (__bf16)f; }

static __device__ __forceinline__ float siluf(float x) { return x / (1.0f + __expf(-x)); }

// hw-transcendental softplus: v_exp_f32 + v_log_f32 instead of log1pf expansion
static __device__ __forceinline__ float softplusf(float v) {
  return (v > 20.0f) ? v : __logf(1.0f + __expf(v));
}

// scan-order t -> original l index for direction n (also used for output un-permute)
static __device__ __forceinline__ int permL(int n, int t) {
  switch (n) {
    case 0:  return t;
    case 1:  return L_ - 1 - t;
    case 2:  return (t & 15) * 16 + (t >> 4);              // (h,w) transpose, H=W=16
    default: { int s = L_ - 1 - t; return (s & 15) * 16 + (s >> 4); }
  }
}

// ---------------------------------------------------------------------------
// Tensor Data Mover: flat 1-D copy of nElems fp32 from global to LDS.
// D# built per CDNA5 ISA:
//   group0: count=1 | lds_addr | global_addr[56:0] | type=2
//   group1: data_size=4B, tensor_dim0 = tile_dim0 = nElems (< 65536)
// Issued by one wave; completion via TENSORcnt.
// ---------------------------------------------------------------------------
static __device__ __forceinline__ void tdm_load_1d(void* lds_dst, const void* gsrc,
                                                   unsigned nElems) {
  unsigned lds = (unsigned)(unsigned long long)lds_dst;     // generic addr low 32 = LDS offset
  unsigned long long ga = (unsigned long long)gsrc;
  uint32x4 g0;
  g0[0] = 1u;                                               // count=1, is_restore=0
  g0[1] = lds;                                              // lds_addr
  g0[2] = (unsigned)(ga & 0xFFFFFFFFull);                   // global_addr[31:0]
  g0[3] = (unsigned)((ga >> 32) & 0x01FFFFFFull) | (2u << 30);  // addr[56:32] | type=2
  int32x8 g1;
  g1[0] = (int)(2u << 16);                                  // data_size=2 (4 bytes)
  g1[1] = (int)((nElems & 0xFFFFu) << 16);                  // tensor_dim0[15:0] @ bit48
  g1[2] = (int)(nElems >> 16);                              // tensor_dim0[31:16]
  g1[3] = (int)((nElems & 0xFFFFu) << 16);                  // tile_dim0 @ bits 127:112
  g1[4] = 0; g1[5] = 0; g1[6] = 0; g1[7] = 0;
  int32x4 z4 = {0, 0, 0, 0};
#if __has_include(<hip/amd_detail/amd_gfx1250_TDM.h>)
  int32x8 z8 = {0, 0, 0, 0, 0, 0, 0, 0};
  __builtin_amdgcn_tensor_load_to_lds(g0, g1, z4, z4, z8, 0);
#else
  __builtin_amdgcn_tensor_load_to_lds(g0, g1, z4, z4, 0);
#endif
}

// ---------------------------------------------------------------------------
// Generic WMMA GEMM: out[q][p] = sum_k w[p][k] * xin[q][k]
//   w   : (P, K) row-major (optionally per-direction: w + (q0/qPerN)*wStrideN)
//   xin : (Q, K) row-major; 16xK fp32 tile pulled to LDS by TDM, converted to
//         bf16 in LDS, B fragment shared by all waves and PT accumulators.
// Each wave: PT p-tiles -> PT v_wmma_f32_16x16x32_bf16 per K-step.
// grid.x = Q/16, grid.y = P/(16*PT*wavesPerBlock)
// ---------------------------------------------------------------------------
template <int PT>
__global__ __launch_bounds__(128) void gemm_bf16_wmma(
    const float* __restrict__ w, const float* __restrict__ xin,
    float* __restrict__ out, int P, int K, int ldo,
    int wStrideN, int qPerN)
{
  __shared__ __align__(32) float  sXf[16 * 512];   // TDM landing zone (fp32), 32 KB
  __shared__ __align__(32) __bf16 sX [16 * 512];   // bf16 operand tile, 16 KB
  const int tid  = threadIdx.x;
  const int lane = tid & 31;
  const int wave = tid >> 5;
  const int nW   = blockDim.x >> 5;
  const int q0   = blockIdx.x * 16;
  const int p0   = (blockIdx.y * nW + wave) * (16 * PT);

  // Async-stage 16 activation rows (16*K contiguous floats) via Tensor Data Mover.
  const float* src = xin + (size_t)q0 * K;
  if (wave == 0) {
    tdm_load_1d(sXf, src, (unsigned)(16 * K));
    __builtin_amdgcn_s_wait_tensorcnt(0);
  }
  __syncthreads();
  // Convert staged tile to bf16 once, cooperatively (v_cvt_pk_bf16_f32).
  for (int i = tid; i < 16 * K; i += blockDim.x) sX[i] = f2bf(sXf[i]);
  __syncthreads();
  if (p0 >= P) return;

  const float* wn = w;
  if (wStrideN) wn += (size_t)(q0 / qPerN) * wStrideN;

  const int hf  = lane >> 4;     // lane half selects K sub-block
  const int l15 = lane & 15;
#pragma unroll
  for (int pt = 0; pt < PT; ++pt)
    __builtin_prefetch(wn + (size_t)(p0 + pt * 16 + l15) * K, 0, 0);

  v8f c[PT];
#pragma unroll
  for (int pt = 0; pt < PT; ++pt) c[pt] = {};

  for (int k0 = 0; k0 < K; k0 += 32) {
    // B fragment (32x16 bf16): column q fixed per lane, 16 contiguous K from LDS
    const v16bf b = *(const v16bf*)&sX[l15 * K + k0 + 16 * hf];
    const int kb = k0 + hf * 16;
#pragma unroll
    for (int pt = 0; pt < PT; ++pt) {
      const float* wp = wn + (size_t)(p0 + pt * 16 + l15) * K;
      v16bf a;   // A fragment (16x32 bf16): 16 contiguous K per lane half
#pragma unroll
      for (int e = 0; e < 16; ++e) {
        a[e]     = f2bf(wp[kb + e]);
      }
      c[pt] = __builtin_amdgcn_wmma_f32_16x16x32_bf16(false, a, false, b,
                                                      (short)0, c[pt], false, false);
    }
  }

  // D: lane holds column q, 8 contiguous p values -> contiguous stores
#pragma unroll
  for (int pt = 0; pt < PT; ++pt) {
    float* op = out + (size_t)(q0 + l15) * ldo + p0 + pt * 16 + 8 * hf;
#pragma unroll
    for (int r = 0; r < 8; ++r) op[r] = c[pt][r];
  }
}

// ---------------------------------------------------------------------------
// dt projection (K=R=16, zero-padded to 32) + softplus, one WMMA per tile
//   dtv[q][d] = softplus( sum_r dbl[q][r] * dt_w[n][d][r] + dt_b[n][d] )
// ---------------------------------------------------------------------------
__global__ __launch_bounds__(128) void dt_softplus_wmma(
    const float* __restrict__ dblv, const float* __restrict__ dt_w,
    const float* __restrict__ dt_b, float* __restrict__ dtv)
{
  const int tid = threadIdx.x, lane = tid & 31, wave = tid >> 5;
  const int q0 = blockIdx.x * 16;                 // over ND*B_*L_
  const int p0 = (blockIdx.y * 4 + wave) * 16;    // over DI
  const int n  = q0 / (B_ * L_);
  const int hf = lane >> 4, l15 = lane & 15;

  // A = dt_w[n] rows d, K = r (0..15 real, 16..31 zero)
  const float* wp = dt_w + (size_t)n * DI * RR + (size_t)(p0 + l15) * RR;
  const int kb = hf * 8;
  v16bf a;
#pragma unroll
  for (int e = 0; e < 8; ++e) { a[e] = f2bf(wp[kb + e]); a[e + 8] = f2bf(0.0f); }

  // B: column q fixed per lane; elements 0..7 carry K=8*hf+e (= dtr), elements 8..15 zero
  const float* xp = dblv + (size_t)(q0 + l15) * 48;
  v16bf b;
#pragma unroll
  for (int e = 0; e < 16; ++e) b[e] = (e < 8) ? f2bf(xp[8 * hf + e]) : f2bf(0.0f);

  v8f c = {};
  c = __builtin_amdgcn_wmma_f32_16x16x32_bf16(false, a, false, b, (short)0, c, false, false);

  const int pb = p0 + 8 * hf;
  float* op = dtv + (size_t)(q0 + l15) * DI + pb;
#pragma unroll
  for (int r = 0; r < 8; ++r)
    op[r] = softplusf(c[r] + dt_b[(size_t)n * DI + pb + r]);
}

// ---------------------------------------------------------------------------
// Causal depthwise conv (K=4) along scan order + SiLU, per direction.
// ---------------------------------------------------------------------------
__global__ __launch_bounds__(512) void conv_silu(
    const float* __restrict__ xz, const float* __restrict__ conv_w,
    const float* __restrict__ conv_b, float* __restrict__ xc)
{
  const int n = blockIdx.x >> 5;
  const int b = blockIdx.x & 31;
  const int d = threadIdx.x;
  const float* cw = conv_w + ((size_t)n * DI + d) * KC;
  const float cw0 = cw[0], cw1 = cw[1], cw2 = cw[2], cw3 = cw[3];
  const float cb  = conv_b[(size_t)n * DI + d];
  float x3 = 0.f, x2 = 0.f, x1 = 0.f;
  const size_t rowBase = ((size_t)n * B_ + b) * L_;
  for (int t = 0; t < L_; ++t) {
    const int l = permL(n, t);
    const float x0 = xz[((size_t)b * L_ + l) * (2 * DI) + d];
    const float s  = cb + cw0 * x3 + cw1 * x2 + cw2 * x1 + cw3 * x0;
    xc[(rowBase + t) * DI + d] = siluf(s);
    x3 = x2; x2 = x1; x1 = x0;
  }
}

// ---------------------------------------------------------------------------
// Selective scan, sequential in t. One block per (n,b); lane = channel d.
// h[16] lives in VGPRs; B_t/C_t broadcast via LDS.
// ---------------------------------------------------------------------------
__global__ __launch_bounds__(512) void selective_scan(
    const float* __restrict__ xz, const float* __restrict__ xc,
    const float* __restrict__ dblv, const float* __restrict__ dtv,
    const float* __restrict__ A_log, const float* __restrict__ D_skip,
    float* __restrict__ outs)
{
  const int n = blockIdx.x >> 5, b = blockIdx.x & 31, d = threadIdx.x;
  __shared__ float sB[NS], sC[NS];
  float Aj[NS], h[NS];
#pragma unroll
  for (int j = 0; j < NS; ++j) {
    Aj[j] = -__expf(A_log[((size_t)n * DI + d) * NS + j]);
    h[j]  = 0.f;
  }
  const float Dsk = D_skip[(size_t)n * DI + d];
  const size_t rowBase = ((size_t)n * B_ + b) * L_;
  for (int t = 0; t < L_; ++t) {
    const size_t q = rowBase + t;
    if (threadIdx.x < 32) {
      float v = dblv[q * 48 + 16 + threadIdx.x];
      if (threadIdx.x < 16) sB[threadIdx.x] = v; else sC[threadIdx.x - 16] = v;
    }
    __syncthreads();
    const float dt = dtv[q * DI + d];
    const float u  = xc[q * DI + d];
    const float du = dt * u;
    float y = 0.f;
#pragma unroll
    for (int j = 0; j < NS; ++j) {
      h[j] = __expf(dt * Aj[j]) * h[j] + du * sB[j];
      y += h[j] * sC[j];
    }
    const int l = permL(n, t);
    const float z = xz[((size_t)b * L_ + l) * (2 * DI) + DI + d];
    outs[(rowBase + l) * DI + d] = (y + u * Dsk) * siluf(z);
    __syncthreads();
  }
}

// ---------------------------------------------------------------------------
// LayerNorm over DI + accumulate mean over L of the normalized values.
// ---------------------------------------------------------------------------
__global__ __launch_bounds__(512) void ln_gmean(
    const float* __restrict__ outs, const float* __restrict__ g_ln,
    const float* __restrict__ b_ln, float* __restrict__ gmean)
{
  const int nb = blockIdx.x, d = threadIdx.x;
  __shared__ float r1[512], r2[512];
  const float gd = g_ln[d], bd = b_ln[d];
  float acc = 0.f;
  for (int l = 0; l < L_; ++l) {
    const float o = outs[((size_t)nb * L_ + l) * DI + d];
    r1[d] = o; r2[d] = o * o;
    __syncthreads();
    for (int s = 256; s > 0; s >>= 1) {
      if (d < s) { r1[d] += r1[d + s]; r2[d] += r2[d + s]; }
      __syncthreads();
    }
    const float mu  = r1[0] * (1.0f / DI);
    const float var = r2[0] * (1.0f / DI) - mu * mu;
    acc += (o - mu) * rsqrtf(var + 1e-5f) * gd + bd;
    __syncthreads();
  }
  gmean[(size_t)nb * DI + d] = acc * (1.0f / L_);
}

// ---------------------------------------------------------------------------
// Gating MLP: g -> gelu(g @ gr_w^T + gr_b) -> sigmoid(. @ cs_w^T + cs_b)
// ---------------------------------------------------------------------------
__global__ __launch_bounds__(512) void gating(
    const float* __restrict__ gmean, const float* __restrict__ gr_w,
    const float* __restrict__ gr_b, const float* __restrict__ cs_w,
    const float* __restrict__ cs_b, float* __restrict__ attn)
{
  const int nb = blockIdx.x, tid = threadIdx.x;
  __shared__ float sg[DI], sr[AR];
  sg[tid] = gmean[(size_t)nb * DI + tid];
  __syncthreads();
  if (tid < AR) {
    float acc = gr_b[tid];
    const float* wr = gr_w + (size_t)tid * DI;
    for (int dd = 0; dd < DI; ++dd) acc += sg[dd] * wr[dd];
    sr[tid] = 0.5f * acc * (1.0f + erff(acc * 0.70710678118654752f));  // exact gelu
  }
  __syncthreads();
  float a = cs_b[tid];
  const float* wc = cs_w + (size_t)tid * AR;
#pragma unroll
  for (int r = 0; r < AR; ++r) a += sr[r] * wc[r];
  attn[(size_t)nb * DI + tid] = 1.0f / (1.0f + __expf(-a));
}

// fused[b][l][d] = sum_n outs[n][b][l][d] * attn[n][b][d]
__global__ __launch_bounds__(256) void fuse_dirs(
    const float* __restrict__ outs, const float* __restrict__ attn,
    float* __restrict__ fused)
{
  const int idx = blockIdx.x * 256 + threadIdx.x;  // over B_*L_*DI
  const int d  = idx % DI;
  const int bl = idx / DI;
  const int b  = bl / L_;
  const int l  = bl % L_;
  float s = 0.f;
#pragma unroll
  for (int n = 0; n < ND; ++n) {
    s += outs[(((size_t)(n * B_ + b)) * L_ + l) * DI + d] *
         attn[(size_t)(n * B_ + b) * DI + d];
  }
  fused[idx] = s;
}

// ---------------------------------------------------------------------------
extern "C" void kernel_launch(void* const* d_in, const int* in_sizes, int n_in,
                              void* d_out, int out_size, void* d_ws, size_t ws_size,
                              hipStream_t stream)
{
  const float* x        = (const float*)d_in[0];
  const float* in_proj  = (const float*)d_in[1];
  const float* conv_w   = (const float*)d_in[2];
  const float* conv_b   = (const float*)d_in[3];
  const float* x_proj_w = (const float*)d_in[4];
  const float* dt_w     = (const float*)d_in[5];
  const float* dt_b     = (const float*)d_in[6];
  const float* A_log    = (const float*)d_in[7];
  const float* D_skip   = (const float*)d_in[8];
  const float* ln_g     = (const float*)d_in[9];
  const float* ln_b     = (const float*)d_in[10];
  const float* gr_w     = (const float*)d_in[11];
  const float* gr_b     = (const float*)d_in[12];
  const float* cs_w     = (const float*)d_in[13];
  const float* cs_b     = (const float*)d_in[14];
  const float* out_w    = (const float*)d_in[15];
  float* out = (float*)d_out;

  char* ws = (char*)d_ws;
  float* xz    = (float*)(ws);                  // (B,L,2DI)          33.55 MB
  float* xc    = (float*)(ws + 33554432);       // (ND,B,L,DI)        67.11 MB
  float* dblv  = (float*)(ws + 100663296);      // (ND,B,L,48)         6.29 MB
  float* dtv   = (float*)(ws + 106954752);      // (ND,B,L,DI)        67.11 MB
  float* outs  = (float*)(ws + 174063616);      // (ND,B,L,DI)        67.11 MB
  float* gmean = (float*)(ws + 241172480);      // (ND,B,DI)           0.26 MB
  float* attn  = (float*)(ws + 241434624);      // (ND,B,DI)           0.26 MB
  float* fused = (float*)(ws + 241696768);      // (B,L,DI)           16.78 MB

  // 1. xz[b,l,e] = sum_d x[b,l,d] * in_proj_w[e,d]  (P=1024,K=256,Q=8192): 4 waves x PT=4
  gemm_bf16_wmma<4><<<dim3(8192 / 16, 1024 / 256), 128, 0, stream>>>(
      in_proj, x, xz, 1024, 256, 2 * DI, 0, 1);

  // 2. per-direction causal conv + SiLU (scan order)
  conv_silu<<<dim3(ND * B_), 512, 0, stream>>>(xz, conv_w, conv_b, xc);

  // 3. dbl[q,e'] = sum_d xc[q,d] * x_proj_w[n][e',d]  (P=48,K=512,Q=32768): 3 waves x PT=1
  gemm_bf16_wmma<1><<<dim3(32768 / 16, 1), 96, 0, stream>>>(
      x_proj_w, xc, dblv, 48, 512, 48, 48 * 512, B_ * L_);

  // 4. dt = softplus(dtr @ dt_w^T + dt_b)
  dt_softplus_wmma<<<dim3(32768 / 16, 512 / 64), 128, 0, stream>>>(dblv, dt_w, dt_b, dtv);

  // 5. selective scan + D-skip + z-gate + un-permute
  selective_scan<<<dim3(ND * B_), 512, 0, stream>>>(xz, xc, dblv, dtv, A_log, D_skip, outs);

  // 6. LayerNorm stats + mean over L
  ln_gmean<<<dim3(ND * B_), 512, 0, stream>>>(outs, ln_g, ln_b, gmean);

  // 7. gating MLP -> per (n,b,d) attention
  gating<<<dim3(ND * B_), 512, 0, stream>>>(gmean, gr_w, gr_b, cs_w, cs_b, attn);

  // 8. fuse directions
  fuse_dirs<<<dim3((B_ * L_ * DI) / 256), 256, 0, stream>>>(outs, attn, fused);

  // 9. out[b,l,o] = sum_d fused[b,l,d] * out_proj_w[o,d]  (P=256,K=512,Q=8192): 1 block-col x PT=4
  gemm_bf16_wmma<4><<<dim3(8192 / 16, 1), 128, 0, stream>>>(
      out_w, fused, out, 256, 512, DM, 0, 1);
}
